// SemSegNet_86165633892441
// MI455X (gfx1250) — compile-verified, matrix operand order checked by reading, not run on hardware
//
#include <hip/hip_runtime.h>
#include <math.h>

// ---------------------------------------------------------------------------
// SemSegNet forward for MI455X (gfx1250, wave32).
// All GEMMs: v_wmma_f32_16x16x32_bf16 with pre-converted, K-padded bf16
// activations and swizzled bf16 weights.
//  - N%128==0 layers: wave computes 16x64 (A fragment reused over 4 WMMAs)
//  - otherwise: wave computes 16x16
// Attention is flash-style, 32 keys/iter so P@V WMMAs run with full K=32.
// ---------------------------------------------------------------------------

#define BB 4
#define NN 4096
#define EPSV 1e-5f

typedef __attribute__((ext_vector_type(16))) __bf16 v16bf;
typedef __attribute__((ext_vector_type(8)))  __bf16 v8bf;
typedef __attribute__((ext_vector_type(8)))  float  v8f;

union bfrag { v16bf v; v8bf h[2]; };

__device__ __forceinline__ __bf16 f2bf(float f) {
  unsigned int u = __float_as_uint(f);
  u += 0x7fffu + ((u >> 16) & 1u);                // round-to-nearest-even
  unsigned short h = (unsigned short)(u >> 16);
  __bf16 r;
  __builtin_memcpy(&r, &h, sizeof(r));
  return r;
}

// ---------------------------------------------------------------------------
// Operand conversion kernels (once per layer; O(MK)+O(KN) vs O(MKN/32) GEMM)
// ---------------------------------------------------------------------------
__global__ void convA(const float* __restrict__ src, __bf16* __restrict__ dst,
                      int K, int Kp, size_t total) {
  size_t t = (size_t)blockIdx.x * blockDim.x + threadIdx.x;
  if (t >= total) return;                     // total = M*Kp
  size_t r = t / Kp;
  int k = (int)(t % Kp);
  dst[t] = (k < K) ? f2bf(src[r * K + k]) : f2bf(0.0f);
}
// fp32 W[K,N] -> bf16 swizzled [Kp/32][N][32], zero-padded
__global__ void convW(const float* __restrict__ W, __bf16* __restrict__ dst,
                      int K, int N, size_t total) {
  size_t t = (size_t)blockIdx.x * blockDim.x + threadIdx.x;
  if (t >= total) return;                     // total = (Kp/32)*N*32
  int j  = (int)(t % 32);
  size_t r = t / 32;
  int n  = (int)(r % N);
  int kt = (int)(r / N);
  int k  = kt * 32 + j;
  dst[t] = (k < K) ? f2bf(W[(size_t)k * N + n]) : f2bf(0.0f);
}
__global__ void conv_scale(const float* __restrict__ src, __bf16* __restrict__ dst,
                           float scale, size_t total) {
  size_t t = (size_t)blockIdx.x * blockDim.x + threadIdx.x;
  if (t < total) dst[t] = f2bf(src[t] * scale);
}
// V [B,N,128] fp32 -> Vt [B,128,N] bf16
__global__ void transpose_conv_v(const float* __restrict__ V, __bf16* __restrict__ Vt,
                                 size_t total) {
  size_t t = (size_t)blockIdx.x * blockDim.x + threadIdx.x;
  if (t >= total) return;                     // total = B*NN*128
  int c = (int)(t % 128);
  size_t r = t / 128;
  int nIdx = (int)(r % NN);
  int b = (int)(r / NN);
  Vt[((size_t)b * 128 + c) * NN + nIdx] = f2bf(V[t]);
}

// ---------------------------------------------------------------------------
// WMMA GEMM (generic):  wave -> one 16x16 tile, block tile 32x32.
// Requires M%32==0, Kp%32==0.
// ---------------------------------------------------------------------------
__global__ void gemm_wmma(const __bf16* __restrict__ A, const __bf16* __restrict__ Bs,
                          const float* __restrict__ bias, float* __restrict__ out,
                          int M, int Kp, int N, int hasBias) {
  const int lane = threadIdx.x & 31;
  const int wave = threadIdx.x >> 5;
  const int rowBase = blockIdx.y * 32 + (wave >> 1) * 16;
  const int colBase = blockIdx.x * 32 + (wave & 1) * 16;
  const int m  = lane & 15;
  const int hf = lane >> 4;
  int cc = colBase + m;
  if (cc > N - 1) cc = N - 1;                 // clamp (discarded at store)

  const __bf16* ap = A + (size_t)(rowBase + m) * Kp + hf * 8;
  const __bf16* bp = Bs + (size_t)cc * 32 + hf * 8;
  const size_t bstep = (size_t)N * 32;

  v8f c;
#pragma unroll
  for (int i = 0; i < 8; ++i) c[i] = 0.0f;

  const int ktn = Kp >> 5;
  for (int kt = 0; kt < ktn; ++kt) {
    bfrag a, b;
    a.h[0] = *(const v8bf*)(ap);
    a.h[1] = *(const v8bf*)(ap + 16);
    b.h[0] = *(const v8bf*)(bp);
    b.h[1] = *(const v8bf*)(bp + 16);
    c = __builtin_amdgcn_wmma_f32_16x16x32_bf16(false, a.v, false, b.v, (short)0, c,
                                                false, false);
    ap += 32;
    bp += bstep;
  }
#pragma unroll
  for (int v = 0; v < 8; ++v) {
    const int r   = rowBase + v + 8 * hf;     // C/D layout: M = vgpr + 8*half
    const int cc0 = colBase + (lane & 15);
    if (cc0 < N) {
      float o = c[v];
      if (hasBias) o += bias[cc0];
      out[(size_t)r * N + cc0] = o;
    }
  }
}

// ---------------------------------------------------------------------------
// WMMA GEMM (wide):  wave -> 16x64 strip (A fragment reused over 4 WMMAs),
// block tile 32x128.  Requires M%32==0, Kp%32==0, N%128==0.
// ---------------------------------------------------------------------------
__global__ void gemm_wmma4(const __bf16* __restrict__ A, const __bf16* __restrict__ Bs,
                           const float* __restrict__ bias, float* __restrict__ out,
                           int M, int Kp, int N, int hasBias) {
  const int lane = threadIdx.x & 31;
  const int wave = threadIdx.x >> 5;
  const int rowBase = blockIdx.y * 32 + (wave >> 1) * 16;
  const int colBase = blockIdx.x * 128 + (wave & 1) * 64;
  const int m  = lane & 15;
  const int hf = lane >> 4;

  const __bf16* ap = A + (size_t)(rowBase + m) * Kp + hf * 8;
  const __bf16* bp = Bs + (size_t)(colBase + m) * 32 + hf * 8;
  const size_t bstep = (size_t)N * 32;

  v8f acc[4];
#pragma unroll
  for (int j = 0; j < 4; ++j)
#pragma unroll
    for (int i = 0; i < 8; ++i) acc[j][i] = 0.0f;

  const int ktn = Kp >> 5;
  for (int kt = 0; kt < ktn; ++kt) {
    bfrag a;
    a.h[0] = *(const v8bf*)(ap);
    a.h[1] = *(const v8bf*)(ap + 16);
#pragma unroll
    for (int j = 0; j < 4; ++j) {
      bfrag b;
      b.h[0] = *(const v8bf*)(bp + j * 512);        // 16 cols x 32 = 512 elems
      b.h[1] = *(const v8bf*)(bp + j * 512 + 16);
      acc[j] = __builtin_amdgcn_wmma_f32_16x16x32_bf16(false, a.v, false, b.v,
                                                       (short)0, acc[j], false, false);
    }
    ap += 32;
    bp += bstep;
  }
#pragma unroll
  for (int j = 0; j < 4; ++j) {
#pragma unroll
    for (int v = 0; v < 8; ++v) {
      const int r  = rowBase + v + 8 * hf;
      const int cc = colBase + j * 16 + (lane & 15);
      float o = acc[j][v];
      if (hasBias) o += bias[cc];
      out[(size_t)r * N + cc] = o;
    }
  }
}

// ---------------------------------------------------------------------------
// BatchNorm over all rows (reference uses current-batch statistics)
// ---------------------------------------------------------------------------
__global__ void colstats(const float* __restrict__ y, int M, int N,
                         float* __restrict__ stats) {
  const int c = blockIdx.x;
  __shared__ float s1[256], s2[256];
  float a = 0.f, b = 0.f;
  for (int r = threadIdx.x; r < M; r += blockDim.x) {
    float v = y[(size_t)r * N + c];
    a += v; b += v * v;
  }
  s1[threadIdx.x] = a; s2[threadIdx.x] = b;
  __syncthreads();
  for (int s = 128; s > 0; s >>= 1) {
    if (threadIdx.x < s) { s1[threadIdx.x] += s1[threadIdx.x + s]; s2[threadIdx.x] += s2[threadIdx.x + s]; }
    __syncthreads();
  }
  if (threadIdx.x == 0) {
    float mean = s1[0] / (float)M;
    stats[c]        = mean;
    stats[1024 + c] = s2[0] / (float)M - mean * mean;
  }
}

__global__ void bn_relu(float* __restrict__ y, const float* __restrict__ stats,
                        const float* __restrict__ g, const float* __restrict__ b,
                        size_t total, int N) {
  size_t t = (size_t)blockIdx.x * blockDim.x + threadIdx.x;
  if (t >= total) return;
  int c = (int)(t % N);
  float v = (y[t] - stats[c]) * rsqrtf(stats[1024 + c] + EPSV) * g[c] + b[c];
  y[t] = fmaxf(v, 0.0f);
}

// ---------------------------------------------------------------------------
// Farthest point sampling: one block per batch, LDS distance array.
// ---------------------------------------------------------------------------
__global__ void fps_kernel(const float* __restrict__ xyz, int n, int npoint,
                           int* __restrict__ idx) {
  const int b = blockIdx.x;
  const float* p = xyz + (size_t)b * n * 3;
  __shared__ float dist[4096];
  __shared__ float rv[256];
  __shared__ int   ri[256];
  const int tid = threadIdx.x, bd = blockDim.x;
  for (int i = tid; i < n; i += bd) dist[i] = 1e10f;
  __syncthreads();
  int far = 0;
  for (int it = 0; it < npoint; ++it) {
    if (tid == 0) idx[b * npoint + it] = far;
    const float fx = p[far * 3 + 0], fy = p[far * 3 + 1], fz = p[far * 3 + 2];
    float best = -1.0f; int bi = 0;
    for (int i = tid; i < n; i += bd) {
      float dx = p[i * 3 + 0] - fx, dy = p[i * 3 + 1] - fy, dz = p[i * 3 + 2] - fz;
      float d = dx * dx + dy * dy + dz * dz;
      float nd = fminf(dist[i], d);
      dist[i] = nd;
      if (nd > best) { best = nd; bi = i; }
    }
    rv[tid] = best; ri[tid] = bi;
    __syncthreads();
    for (int s = bd / 2; s > 0; s >>= 1) {
      if (tid < s) {
        if (rv[tid + s] > rv[tid] ||
            (rv[tid + s] == rv[tid] && ri[tid + s] < ri[tid])) {
          rv[tid] = rv[tid + s]; ri[tid] = ri[tid + s];
        }
      }
      __syncthreads();
    }
    far = ri[0];
    __syncthreads();
  }
}

__global__ void gather_xyz(const float* __restrict__ xyz, const int* __restrict__ idx,
                           int n, int S, float* __restrict__ out, int total) {
  int t = blockIdx.x * blockDim.x + threadIdx.x;
  if (t >= total) return;
  int b = t / S;
  int j = idx[t];
  out[(size_t)t * 3 + 0] = xyz[((size_t)b * n + j) * 3 + 0];
  out[(size_t)t * 3 + 1] = xyz[((size_t)b * n + j) * 3 + 1];
  out[(size_t)t * 3 + 2] = xyz[((size_t)b * n + j) * 3 + 2];
}

// Ball query: ascending-index membership, pad with first hit.
__global__ void ballq(const float* __restrict__ xyz, const float* __restrict__ nxyz,
                      int n, int S, float r2, int ns, int* __restrict__ out, int total) {
  int t = blockIdx.x * blockDim.x + threadIdx.x;
  if (t >= total) return;
  int b = t / S;
  const float cx = nxyz[(size_t)t * 3 + 0];
  const float cy = nxyz[(size_t)t * 3 + 1];
  const float cz = nxyz[(size_t)t * 3 + 2];
  int* o = out + (size_t)t * ns;
  int cnt = 0, first = 0;
  for (int j = 0; j < n && cnt < ns; ++j) {
    float dx = xyz[((size_t)b * n + j) * 3 + 0] - cx;
    float dy = xyz[((size_t)b * n + j) * 3 + 1] - cy;
    float dz = xyz[((size_t)b * n + j) * 3 + 2] - cz;
    if (dx * dx + dy * dy + dz * dz <= r2) {
      if (cnt == 0) first = j;
      o[cnt++] = j;
    }
  }
  for (; cnt < ns; ++cnt) o[cnt] = first;
}

// grouped = concat([xyz[gidx]-new_xyz, pts[gidx]])
__global__ void group_kernel(const float* __restrict__ xyz, const float* __restrict__ nxyz,
                             const float* __restrict__ pts, int Cp,
                             const int* __restrict__ gidx, int S, int K, int n,
                             float* __restrict__ out, int total) {
  int t = blockIdx.x * blockDim.x + threadIdx.x;
  if (t >= total) return;                    // total = B*S*K
  int bs = t / K;
  int b = bs / S;
  int j = gidx[t];
  float* o = out + (size_t)t * (3 + Cp);
  o[0] = xyz[((size_t)b * n + j) * 3 + 0] - nxyz[(size_t)bs * 3 + 0];
  o[1] = xyz[((size_t)b * n + j) * 3 + 1] - nxyz[(size_t)bs * 3 + 1];
  o[2] = xyz[((size_t)b * n + j) * 3 + 2] - nxyz[(size_t)bs * 3 + 2];
  for (int c = 0; c < Cp; ++c) o[3 + c] = pts[((size_t)b * n + j) * Cp + c];
}

// SA3 group-all: concat([l2_xyz, l2_pts]) (no recentering)
__global__ void groupall3(const float* __restrict__ xyz, const float* __restrict__ pts,
                          float* __restrict__ out, int total) {
  int t = blockIdx.x * blockDim.x + threadIdx.x;
  if (t >= total) return;                    // total = B*128
  float* o = out + (size_t)t * 259;
  o[0] = xyz[(size_t)t * 3 + 0];
  o[1] = xyz[(size_t)t * 3 + 1];
  o[2] = xyz[(size_t)t * 3 + 2];
  for (int c = 0; c < 256; ++c) o[3 + c] = pts[(size_t)t * 256 + c];
}

// Attention pooling: out[bs,c] = sum_k softmax_k(h.att_w + att_b) * h[bs,k,c]
__global__ void att_pool(const float* __restrict__ h, const float* __restrict__ aw,
                         const float* __restrict__ ab, float* __restrict__ out,
                         int K, int C) {
  const int bs = blockIdx.x;
  const int tid = threadIdx.x;
  __shared__ float sw[128], red[128];
  const float* hb = h + (size_t)bs * K * C;
  float sc = -1e30f;
  if (tid < K) {
    sc = ab[0];
    for (int c = 0; c < C; ++c) sc += hb[(size_t)tid * C + c] * aw[c];
  }
  sw[tid] = sc;
  __syncthreads();
  red[tid] = sw[tid];
  __syncthreads();
  for (int s = 64; s > 0; s >>= 1) {
    if (tid < s) red[tid] = fmaxf(red[tid], red[tid + s]);
    __syncthreads();
  }
  const float mx = red[0];
  __syncthreads();
  float e = (tid < K) ? __expf(sw[tid] - mx) : 0.0f;
  red[tid] = e;
  __syncthreads();
  for (int s = 64; s > 0; s >>= 1) {
    if (tid < s) red[tid] += red[tid + s];
    __syncthreads();
  }
  const float sum = red[0];
  __syncthreads();
  sw[tid] = e / sum;
  __syncthreads();
  for (int c = tid; c < C; c += blockDim.x) {
    float acc = 0.f;
    for (int k = 0; k < K; ++k) acc += hb[(size_t)k * C + c] * sw[k];
    out[(size_t)bs * C + c] = acc;
  }
}

// ---------------------------------------------------------------------------
// SE block pieces
// ---------------------------------------------------------------------------
__global__ void se_mean(const float* __restrict__ x, int Np, int C,
                        float* __restrict__ out, int total) {
  int t = blockIdx.x * blockDim.x + threadIdx.x;
  if (t >= total) return;                    // total = B*C
  int b = t / C, c = t % C;
  float a = 0.f;
  for (int i = 0; i < Np; ++i) a += x[((size_t)b * Np + i) * C + c];
  out[t] = a / (float)Np;
}
__global__ void se_fc1(const float* __restrict__ m, const float* __restrict__ fc1,
                       int C, int H, float* __restrict__ out, int total) {
  int t = blockIdx.x * blockDim.x + threadIdx.x;
  if (t >= total) return;                    // total = B*H
  int b = t / H, h = t % H;
  float a = 0.f;
  for (int c = 0; c < C; ++c) a += m[(size_t)b * C + c] * fc1[(size_t)c * H + h];
  out[t] = fmaxf(a, 0.0f);
}
__global__ void se_fc2(const float* __restrict__ hid, const float* __restrict__ fc2,
                       int H, int C, float* __restrict__ out, int total) {
  int t = blockIdx.x * blockDim.x + threadIdx.x;
  if (t >= total) return;                    // total = B*C
  int b = t / C, c = t % C;
  float a = 0.f;
  for (int h = 0; h < H; ++h) a += hid[(size_t)b * H + h] * fc2[(size_t)h * C + c];
  out[t] = 1.0f / (1.0f + __expf(-a));
}
__global__ void se_scale(const float* __restrict__ x, const float* __restrict__ gate,
                         int Np, int C, float* __restrict__ dst, size_t total) {
  size_t t = (size_t)blockIdx.x * blockDim.x + threadIdx.x;
  if (t >= total) return;
  int c = (int)(t % C);
  int b = (int)(t / ((size_t)Np * C));
  dst[t] = x[t] * gate[(size_t)b * C + c];
}

// ---------------------------------------------------------------------------
// Feature propagation helpers
// ---------------------------------------------------------------------------
__global__ void copy2d(const float* __restrict__ src, float* __restrict__ dst,
                       int C, int ld, int coff, size_t total) {
  size_t t = (size_t)blockIdx.x * blockDim.x + threadIdx.x;
  if (t >= total) return;                    // total = M*C
  size_t r = t / C; int c = (int)(t % C);
  dst[r * ld + coff + c] = src[t];
}
__global__ void broadcast_rows(const float* __restrict__ src, float* __restrict__ dst,
                               int rowsPerB, int C, int ld, int coff, size_t total) {
  size_t t = (size_t)blockIdx.x * blockDim.x + threadIdx.x;
  if (t >= total) return;                    // total = B*rowsPerB*C
  int c = (int)(t % C);
  size_t r = t / C;
  int b = (int)(r / rowsPerB);
  dst[r * ld + coff + c] = src[(size_t)b * C + c];
}
__global__ void three_interp(const float* __restrict__ xyz1, const float* __restrict__ xyz2,
                             const float* __restrict__ pts2, int n1, int n2, int C,
                             float* __restrict__ dst, int ld, int coff, int total) {
  int t = blockIdx.x * blockDim.x + threadIdx.x;
  if (t >= total) return;                    // total = B*n1
  int b = t / n1;
  const float px = xyz1[(size_t)t * 3 + 0];
  const float py = xyz1[(size_t)t * 3 + 1];
  const float pz = xyz1[(size_t)t * 3 + 2];
  float d0 = 1e10f, d1 = 1e10f, d2 = 1e10f;
  int i0 = 0, i1 = 0, i2 = 0;
  for (int j = 0; j < n2; ++j) {
    float dx = xyz2[((size_t)b * n2 + j) * 3 + 0] - px;
    float dy = xyz2[((size_t)b * n2 + j) * 3 + 1] - py;
    float dz = xyz2[((size_t)b * n2 + j) * 3 + 2] - pz;
    float d = dx * dx + dy * dy + dz * dz;
    if (d < d0)      { d2 = d1; i2 = i1; d1 = d0; i1 = i0; d0 = d; i0 = j; }
    else if (d < d1) { d2 = d1; i2 = i1; d1 = d; i1 = j; }
    else if (d < d2) { d2 = d; i2 = j; }
  }
  float w0 = 1.0f / (d0 + 1e-8f), w1 = 1.0f / (d1 + 1e-8f), w2 = 1.0f / (d2 + 1e-8f);
  float ws = w0 + w1 + w2;
  w0 /= ws; w1 /= ws; w2 /= ws;
  float* o = dst + (size_t)t * ld + coff;
  const float* pb = pts2 + (size_t)b * n2 * C;
  for (int c = 0; c < C; ++c)
    o[c] = w0 * pb[(size_t)i0 * C + c] + w1 * pb[(size_t)i1 * C + c] + w2 * pb[(size_t)i2 * C + c];
}
__global__ void lin_interp_k(const float* __restrict__ src, int s, int nout, int C,
                             float* __restrict__ dst, int ld, int coff, size_t total) {
  size_t t = (size_t)blockIdx.x * blockDim.x + threadIdx.x;
  if (t >= total) return;                    // total = B*nout*C
  int c = (int)(t % C);
  size_t r = t / C;
  int n = (int)(r % nout);
  int b = (int)(r / nout);
  float pos = (float)n * (float)(s - 1) / (float)(nout - 1);
  int a0 = (int)floorf(pos);
  if (a0 > s - 1) a0 = s - 1;
  if (a0 < 0) a0 = 0;
  int a1 = a0 + 1; if (a1 > s - 1) a1 = s - 1;
  float f = pos - (float)a0;
  dst[r * ld + coff + c] = src[((size_t)b * s + a0) * C + c] * (1.0f - f) +
                           src[((size_t)b * s + a1) * C + c] * f;
}

__global__ void add_inplace(float* __restrict__ a, const float* __restrict__ b, size_t total) {
  size_t t = (size_t)blockIdx.x * blockDim.x + threadIdx.x;
  if (t < total) a[t] += b[t];
}
__global__ void split_xyz(const float* __restrict__ x, float* __restrict__ out, size_t total) {
  size_t t = (size_t)blockIdx.x * blockDim.x + threadIdx.x;
  if (t >= total) return;                    // total = B*N*3
  size_t r = t / 3; int d = (int)(t % 3);
  out[t] = x[r * 6 + d];
}

// ---------------------------------------------------------------------------
// Flash attention: feat = l0_pts + gamma * softmax(q k^T / 4) v
// Q,K bf16 (scale folded into Q); V pre-transposed [B,128,N] bf16.
// One wave per 16-query tile; 32 keys per iteration:
//   2 S-WMMAs (K=16 padded) + 8 P@V WMMAs at FULL K=32 (P staged in LDS bf16).
// ---------------------------------------------------------------------------
#define ATT_WAVES 4
__global__ void flash_attn(const __bf16* __restrict__ Q, const __bf16* __restrict__ Km,
                           const __bf16* __restrict__ Vt, const float* __restrict__ l0p,
                           const float* __restrict__ gammap, float* __restrict__ feat,
                           int n) {
  __shared__ __bf16 pshm[ATT_WAVES * 512];    // 16x32 bf16 tile per wave
  const int lane = threadIdx.x & 31;
  const int wave = threadIdx.x >> 5;
  const int qt   = blockIdx.x * ATT_WAVES + wave;
  const int tilesPerB = n / 16;
  const int b    = qt / tilesPerB;
  const int rowBase = (qt % tilesPerB) * 16;
  const int m  = lane & 15;
  const int hf = lane >> 4;
  __bf16* pp = pshm + wave * 512;

  v8bf z8;
#pragma unroll
  for (int i = 0; i < 8; ++i) z8[i] = f2bf(0.0f);

  // q fragment: 8 real features (hf selects group), upper half zero-pad
  bfrag aq;
  aq.h[0] = *(const v8bf*)(Q + ((size_t)b * n + rowBase + m) * 16 + hf * 8);
  aq.h[1] = z8;

  // per-row V pointers (col = j*16 + m), advanced via kb offset
  const __bf16* vp[8];
#pragma unroll
  for (int j = 0; j < 8; ++j)
    vp[j] = Vt + ((size_t)b * 128 + j * 16 + m) * n + hf * 8;

  float mrow[8], lrow[8], alpha[8];
  v8f o[8];
#pragma unroll
  for (int v = 0; v < 8; ++v) {
    mrow[v] = -1e30f; lrow[v] = 0.0f;
#pragma unroll
    for (int j = 0; j < 8; ++j) o[v][j] = 0.0f;
  }

  const int kiters = n / 32;
  for (int kt = 0; kt < kiters; ++kt) {
    const int kb = kt * 32;
    bfrag bk0, bk1;
    bk0.h[0] = *(const v8bf*)(Km + ((size_t)b * n + kb + m) * 16 + hf * 8);
    bk0.h[1] = z8;
    bk1.h[0] = *(const v8bf*)(Km + ((size_t)b * n + kb + 16 + m) * 16 + hf * 8);
    bk1.h[1] = z8;
    v8f zc;
#pragma unroll
    for (int i = 0; i < 8; ++i) zc[i] = 0.0f;
    v8f s0 = __builtin_amdgcn_wmma_f32_16x16x32_bf16(false, aq.v, false, bk0.v,
                                                     (short)0, zc, false, false);
    v8f s1 = __builtin_amdgcn_wmma_f32_16x16x32_bf16(false, aq.v, false, bk1.v,
                                                     (short)0, zc, false, false);
    // online softmax over 32 keys; row (v + 8*hf) spans the 16 lanes of a half
#pragma unroll
    for (int v = 0; v < 8; ++v) {
      float sv0 = s0[v], sv1 = s1[v];
      float mx = fmaxf(sv0, sv1);
      for (int off = 1; off < 16; off <<= 1) mx = fmaxf(mx, __shfl_xor(mx, off, 16));
      float mN = fmaxf(mrow[v], mx);
      float p0 = __expf(sv0 - mN);
      float p1 = __expf(sv1 - mN);
      float ps = p0 + p1;
      for (int off = 1; off < 16; off <<= 1) ps += __shfl_xor(ps, off, 16);
      float al = __expf(mrow[v] - mN);
      lrow[v] = lrow[v] * al + ps;
      mrow[v] = mN;
      alpha[v] = al;
      const int row = (v + 8 * hf) * 32;
      pp[row + m]      = f2bf(p0);
      pp[row + 16 + m] = f2bf(p1);
    }
    __syncthreads();
    bfrag pa;                                  // full 16x32 P tile (no padding)
    pa.h[0] = *(const v8bf*)(pp + m * 32 + hf * 8);
    pa.h[1] = *(const v8bf*)(pp + m * 32 + 16 + hf * 8);
    __syncthreads();
#pragma unroll
    for (int j = 0; j < 8; ++j) {
      bfrag bv;                                // full 32x16 V tile
      bv.h[0] = *(const v8bf*)(vp[j] + kb);
      bv.h[1] = *(const v8bf*)(vp[j] + kb + 16);
      v8f oj = o[j];
#pragma unroll
      for (int v = 0; v < 8; ++v) oj[v] *= alpha[v];
      o[j] = __builtin_amdgcn_wmma_f32_16x16x32_bf16(false, pa.v, false, bv.v,
                                                     (short)0, oj, false, false);
    }
  }
  const float g = gammap[0];
#pragma unroll
  for (int j = 0; j < 8; ++j) {
#pragma unroll
    for (int v = 0; v < 8; ++v) {
      int r = rowBase + v + 8 * hf;
      int c = j * 16 + m;
      size_t idx = ((size_t)b * n + r) * 128 + c;
      feat[idx] = l0p[idx] + g * (o[j][v] / lrow[v]);
    }
  }
}

// ---------------------------------------------------------------------------
// Host orchestration
// ---------------------------------------------------------------------------
static inline unsigned nb(size_t total) { return (unsigned)((total + 255) / 256); }

extern "C" void kernel_launch(void* const* d_in, const int* in_sizes, int n_in,
                              void* d_out, int out_size, void* d_ws, size_t ws_size,
                              hipStream_t stream) {
  (void)in_sizes; (void)n_in; (void)out_size; (void)ws_size;
  auto P = [&](int i) -> const float* { return (const float*)d_in[i]; };

  // Param indices (dict insertion order of setup_inputs / _init_params):
  // 0:x | sa1: 1-12 mlp(W,b,g,bb)x3, 13 att_w, 14 att_b
  // sa2: 15-26, 27, 28 | sa3: 29-40, 41, 42
  // se_global: 43 fc1, 44 fc2 | se_mid: 45, 46
  // fp3: 47-54 mlp x2, 55 fc1, 56 fc2 | fp2: 57-68 mlp x3, 69, 70
  // fp1: 71-82 mlp x3, 83, 84
  // pa: 85 pos_w, 86 pos_b, 87 wq, 88 wk, 89 wv, 90 gamma
  // fuse: 91-94 | head: 95-106 mlp x3, 107 out_w, 108 out_b
  const float* x = P(0);

  float* ws = (float*)d_ws;
  size_t off = 0;
  auto alloc = [&](size_t nfl) -> float* {
    float* p = ws + off;
    off += (nfl + 63) & ~(size_t)63;
    return p;
  };
  float* l0xyz = alloc((size_t)BB * NN * 3);
  int*   fidx1 = (int*)alloc((size_t)BB * 512);
  int*   gidx1 = (int*)alloc((size_t)BB * 512 * 32);
  float* l1xyz = alloc((size_t)BB * 512 * 3);
  int*   fidx2 = (int*)alloc((size_t)BB * 128);
  int*   gidx2 = (int*)alloc((size_t)BB * 128 * 64);
  float* l2xyz = alloc((size_t)BB * 128 * 3);
  float* BUF0  = alloc((size_t)16384 * 1408);
  float* BUF1  = alloc((size_t)65536 * 128);
  float* BUF2  = alloc((size_t)65536 * 128);
  float* l1pts = alloc((size_t)BB * 512 * 128);
  float* l2pts = alloc((size_t)BB * 128 * 256);
  float* l3pts = alloc((size_t)BB * 1024);
  float* l0pts = alloc((size_t)BB * NN * 128);
  float* featb = alloc((size_t)BB * NN * 128);
  float* hpa   = alloc((size_t)BB * NN * 128);
  float* qb    = alloc((size_t)BB * NN * 16);
  float* kb    = alloc((size_t)BB * NN * 16);
  float* vb    = alloc((size_t)BB * NN * 128);
  float* stats = alloc(2048);
  float* sem   = alloc((size_t)BB * 1024);
  float* seh   = alloc((size_t)BB * 64);
  float* seg   = alloc((size_t)BB * 1024);
  // bf16 scratch (element counts halved into float words)
  __bf16* Abf  = (__bf16*)alloc((size_t)16384 * 1408 / 2);   // max M*Kp
  __bf16* Wswz = (__bf16*)alloc((size_t)1048576 / 2);        // max Kp*N
  __bf16* Qbf  = (__bf16*)alloc((size_t)BB * NN * 16 / 2);
  __bf16* Kbf  = (__bf16*)alloc((size_t)BB * NN * 16 / 2);
  __bf16* Vtb  = (__bf16*)alloc((size_t)BB * NN * 128 / 2);

  // GEMM with operand conversion: out = in[M,K] @ W[K,N] (+bias)
  auto gemmf = [&](const float* in, const float* W, const float* bias, float* out,
                   int M, int K, int N, bool hb) {
    const int Kp = (K + 31) & ~31;
    convA<<<nb((size_t)M * Kp), 256, 0, stream>>>(in, Abf, K, Kp, (size_t)M * Kp);
    convW<<<nb((size_t)Kp * N), 256, 0, stream>>>(W, Wswz, K, N, (size_t)Kp * N);
    if ((N & 127) == 0) {
      dim3 g(N / 128, M / 32);
      gemm_wmma4<<<g, 128, 0, stream>>>(Abf, Wswz, bias, out, M, Kp, N, hb ? 1 : 0);
    } else {
      dim3 g((N + 31) / 32, M / 32);
      gemm_wmma<<<g, 128, 0, stream>>>(Abf, Wswz, bias, out, M, Kp, N, hb ? 1 : 0);
    }
  };
  auto mlayer = [&](const float* in, int wi, float* out, int M, int K, int N) {
    gemmf(in, P(wi), P(wi + 1), out, M, K, N, true);
    colstats<<<N, 256, 0, stream>>>(out, M, N, stats);
    size_t tot = (size_t)M * N;
    bn_relu<<<nb(tot), 256, 0, stream>>>(out, stats, P(wi + 2), P(wi + 3), tot, N);
  };
  auto seblk = [&](const float* src, int Np, int C, int H, const float* fc1,
                   const float* fc2, float* dst) {
    se_mean<<<nb((size_t)BB * C), 256, 0, stream>>>(src, Np, C, sem, BB * C);
    se_fc1<<<nb((size_t)BB * H), 256, 0, stream>>>(sem, fc1, C, H, seh, BB * H);
    se_fc2<<<nb((size_t)BB * C), 256, 0, stream>>>(seh, fc2, H, C, seg, BB * C);
    size_t tot = (size_t)BB * Np * C;
    se_scale<<<nb(tot), 256, 0, stream>>>(src, seg, Np, C, dst, tot);
  };

  // ---- l0 ----
  split_xyz<<<nb((size_t)BB * NN * 3), 256, 0, stream>>>(x, l0xyz, (size_t)BB * NN * 3);

  // ---- SA1 ----
  fps_kernel<<<BB, 256, 0, stream>>>(l0xyz, NN, 512, fidx1);
  gather_xyz<<<nb(BB * 512), 256, 0, stream>>>(l0xyz, fidx1, NN, 512, l1xyz, BB * 512);
  ballq<<<nb(BB * 512), 256, 0, stream>>>(l0xyz, l1xyz, NN, 512, 0.04f, 32, gidx1, BB * 512);
  group_kernel<<<nb((size_t)BB * 512 * 32), 256, 0, stream>>>(
      l0xyz, l1xyz, x, 6, gidx1, 512, 32, NN, BUF0, BB * 512 * 32);
  mlayer(BUF0, 1, BUF1, 65536, 9, 64);
  mlayer(BUF1, 5, BUF2, 65536, 64, 64);
  mlayer(BUF2, 9, BUF1, 65536, 64, 128);
  att_pool<<<BB * 512, 128, 0, stream>>>(BUF1, P(13), P(14), l1pts, 32, 128);

  // ---- SA2 ----
  fps_kernel<<<BB, 256, 0, stream>>>(l1xyz, 512, 128, fidx2);
  gather_xyz<<<nb(BB * 128), 256, 0, stream>>>(l1xyz, fidx2, 512, 128, l2xyz, BB * 128);
  ballq<<<nb(BB * 128), 256, 0, stream>>>(l1xyz, l2xyz, 512, 128, 0.16f, 64, gidx2, BB * 128);
  group_kernel<<<nb((size_t)BB * 128 * 64), 256, 0, stream>>>(
      l1xyz, l2xyz, l1pts, 128, gidx2, 128, 64, 512, BUF0, BB * 128 * 64);
  mlayer(BUF0, 15, BUF1, 32768, 131, 128);
  mlayer(BUF1, 19, BUF2, 32768, 128, 128);
  mlayer(BUF2, 23, BUF1, 32768, 128, 256);
  att_pool<<<BB * 128, 128, 0, stream>>>(BUF1, P(27), P(28), l2pts, 64, 256);

  // ---- SA3 (group all) ----
  groupall3<<<nb(BB * 128), 256, 0, stream>>>(l2xyz, l2pts, BUF0, BB * 128);
  mlayer(BUF0, 29, BUF1, 512, 259, 256);
  mlayer(BUF1, 33, BUF2, 512, 256, 512);
  mlayer(BUF2, 37, BUF1, 512, 512, 1024);
  att_pool<<<BB, 128, 0, stream>>>(BUF1, P(41), P(42), l3pts, 128, 1024);

  // ---- SE global on l3 ----
  seblk(l3pts, 1, 1024, 64, P(43), P(44), l3pts);

  // ---- FP3: l2 <- (l2_pts, broadcast l3_pts) ----
  copy2d<<<nb((size_t)BB * 128 * 256), 256, 0, stream>>>(l2pts, BUF0, 256, 1280, 0,
                                                         (size_t)BB * 128 * 256);
  broadcast_rows<<<nb((size_t)BB * 128 * 1024), 256, 0, stream>>>(
      l3pts, BUF0, 128, 1024, 1280, 256, (size_t)BB * 128 * 1024);
  mlayer(BUF0, 47, BUF1, 512, 1280, 256);
  mlayer(BUF1, 51, BUF2, 512, 256, 256);
  seblk(BUF2, 128, 256, 64, P(55), P(56), l2pts);
  // ---- SE mid ----
  seblk(l2pts, 128, 256, 32, P(45), P(46), l2pts);

  // ---- FP2: l1 <- (l1_pts, 3NN interp of l2_pts) ----
  copy2d<<<nb((size_t)BB * 512 * 128), 256, 0, stream>>>(l1pts, BUF0, 128, 384, 0,
                                                         (size_t)BB * 512 * 128);
  three_interp<<<nb(BB * 512), 256, 0, stream>>>(l1xyz, l2xyz, l2pts, 512, 128, 256,
                                                 BUF0, 384, 128, BB * 512);
  mlayer(BUF0, 57, BUF1, 2048, 384, 256);
  mlayer(BUF1, 61, BUF2, 2048, 256, 256);
  mlayer(BUF2, 65, BUF1, 2048, 256, 128);
  seblk(BUF1, 512, 128, 32, P(69), P(70), l1pts);

  // ---- FP1: l0 <- (x, 3NN interp of l1_pts) ----
  copy2d<<<nb((size_t)BB * NN * 6), 256, 0, stream>>>(x, BUF0, 6, 134, 0,
                                                      (size_t)BB * NN * 6);
  three_interp<<<nb(BB * NN), 256, 0, stream>>>(l0xyz, l1xyz, l1pts, NN, 512, 128,
                                                BUF0, 134, 6, BB * NN);
  mlayer(BUF0, 71, BUF1, 16384, 134, 128);
  mlayer(BUF1, 75, BUF2, 16384, 128, 128);
  mlayer(BUF2, 79, BUF1, 16384, 128, 128);
  seblk(BUF1, NN, 128, 32, P(83), P(84), l0pts);

  // ---- Point attention ----
  gemmf(l0xyz, P(85), P(86), hpa, BB * NN, 3, 128, true);    // pos embed + bias
  add_inplace<<<nb((size_t)BB * NN * 128), 256, 0, stream>>>(hpa, l0pts,
                                                             (size_t)BB * NN * 128);
  gemmf(hpa, P(87), nullptr, qb, BB * NN, 128, 16, false);
  gemmf(hpa, P(88), nullptr, kb, BB * NN, 128, 16, false);
  gemmf(hpa, P(89), nullptr, vb, BB * NN, 128, 128, false);
  conv_scale<<<nb((size_t)BB * NN * 16), 256, 0, stream>>>(qb, Qbf, 0.25f,
                                                           (size_t)BB * NN * 16);
  conv_scale<<<nb((size_t)BB * NN * 16), 256, 0, stream>>>(kb, Kbf, 1.0f,
                                                           (size_t)BB * NN * 16);
  transpose_conv_v<<<nb((size_t)BB * NN * 128), 256, 0, stream>>>(vb, Vtb,
                                                                  (size_t)BB * NN * 128);
  flash_attn<<<(BB * (NN / 16)) / ATT_WAVES, 32 * ATT_WAVES, 0, stream>>>(
      Qbf, Kbf, Vtb, l0pts, P(90), featb, NN);

  // ---- Fuse: concat [feat(128) | global(1024) | lin_interp(256)] ----
  copy2d<<<nb((size_t)BB * NN * 128), 256, 0, stream>>>(featb, BUF0, 128, 1408, 0,
                                                        (size_t)BB * NN * 128);
  broadcast_rows<<<nb((size_t)BB * NN * 1024), 256, 0, stream>>>(
      l3pts, BUF0, NN, 1024, 1408, 128, (size_t)BB * NN * 1024);
  lin_interp_k<<<nb((size_t)BB * NN * 256), 256, 0, stream>>>(
      l2pts, 128, NN, 256, BUF0, 1408, 1152, (size_t)BB * NN * 256);
  mlayer(BUF0, 91, BUF1, 16384, 1408, 256);

  // ---- Head ----
  mlayer(BUF1, 95, BUF2, 16384, 256, 256);
  mlayer(BUF2, 99, BUF1, 16384, 256, 128);
  mlayer(BUF1, 103, BUF2, 16384, 128, 64);
  gemmf(BUF2, P(107), P(108), (float*)d_out, 16384, 64, 11, true);
}
